// acLSTMQuat_36146444763894
// MI455X (gfx1250) — matrix operand order repaired and checked by reading, not compiled, not run on hardware
//
#include <hip/hip_runtime.h>

#define DEVINL __device__ __forceinline__

typedef __attribute__((ext_vector_type(16))) __bf16 bf16x16;
typedef __attribute__((ext_vector_type(8)))  __bf16 bf16x8;
typedef __attribute__((ext_vector_type(8)))  float  v8f;

// ---------------- problem constants ----------------
constexpr int B    = 32;
constexpr int Hd   = 1024;
constexpr int F    = 175;
constexpr int SEED = 100;
constexpr int G4H  = 4 * Hd;           // 4096 gate columns per layer
constexpr int KX   = 192;              // F padded to multiple of 32
constexpr int NT_DEC = (F + 15) / 16;  // 11 decoder column tiles

constexpr int NWG   = 52;              // persistent grid (must be co-resident)
constexpr int BLOCK = 128;             // 4 waves/WG -> 208 waves
constexpr int NGATEW = 192;            // 3 layers x 64 unit-tiles
constexpr int DECW0  = 192;            // decoder waves 192..202

// ---------------- workspace layout (bytes) ----------------
constexpr size_t alignup(size_t x) { return (x + 255) & ~(size_t)255; }
constexpr size_t OFF_BAR   = 0;                         // [0]=cnt [16]=gen [32]=flag
constexpr size_t OFF_WIH1  = 256;
constexpr size_t SZ_WIH1   = (size_t)G4H * KX * 2;      // bf16 [4096][192] (zero-padded cols)
constexpr size_t OFF_WHH1  = OFF_WIH1 + alignup(SZ_WIH1);
constexpr size_t SZ_WHH    = (size_t)G4H * Hd * 2;      // bf16 [4096][1024]
constexpr size_t OFF_WIH2  = OFF_WHH1 + alignup(SZ_WHH);
constexpr size_t OFF_WHH2  = OFF_WIH2 + alignup(SZ_WHH);
constexpr size_t OFF_WIH3  = OFF_WHH2 + alignup(SZ_WHH);
constexpr size_t OFF_WHH3  = OFF_WIH3 + alignup(SZ_WHH);
constexpr size_t OFF_WDEC  = OFF_WHH3 + alignup(SZ_WHH);
constexpr size_t SZ_WDEC   = (size_t)176 * Hd * 2;      // bf16 [176][1024], row 175 = 0
constexpr size_t OFF_BSUM  = OFF_WDEC + alignup(SZ_WDEC);
constexpr size_t SZ_BSUM   = (size_t)3 * G4H * 4;       // f32 b_ih+b_hh per layer
constexpr size_t OFF_H     = OFF_BSUM + alignup(SZ_BSUM);
constexpr size_t SZ_H      = (size_t)2 * 3 * B * Hd * 2; // bf16 ping-pong h[2][3][32][1024]
constexpr size_t OFF_X     = OFF_H + alignup(SZ_H);
constexpr size_t SZ_X      = (size_t)2 * B * KX * 2;     // bf16 ping-pong x[2][32][192]

// ---------------- WMMA fragment loaders ----------------
// A 16x32 bf16: lane (l&15)=row; lanes<16 hold K {k0..k0+15},
// lanes>=16 hold K {k0+16..k0+31}; two contiguous 8-half runs.
DEVINL bf16x16 load_afrag(const __bf16* A, int lda, int mtile, int k0, int lane) {
    int row = (lane & 15) + mtile * 16;
    int hi  = lane >> 4;
    const __bf16* p = A + (size_t)row * lda + k0 + hi * 16;
    bf16x8 lo = *reinterpret_cast<const bf16x8*>(p);
    bf16x8 hh = *reinterpret_cast<const bf16x8*>(p + 8);
    return __builtin_shufflevector(lo, hh, 0,1,2,3,4,5,6,7,8,9,10,11,12,13,14,15);
}

DEVINL v8f wmma_bf16(bf16x16 a, bf16x16 b, v8f c) {
    return __builtin_amdgcn_wmma_f32_16x16x32_bf16(false, a, false, b, (short)0, c, false, false);
}

// Decoder: one [32x16] tile of A[32xK] * W[16 rows x K]^T
DEVINL void gemm_acc2(const __bf16* A, int K, const __bf16* W, int ntile, int lane,
                      v8f& acc0, v8f& acc1) {
    const __bf16* wp = W + (size_t)(ntile * 16 + (lane & 15)) * K + (lane >> 4) * 16;
    for (int k0 = 0; k0 < K; k0 += 32) {
        bf16x16 b  = *reinterpret_cast<const bf16x16*>(wp + k0);
        bf16x16 a0 = load_afrag(A, K, 0, k0, lane);
        bf16x16 a1 = load_afrag(A, K, 1, k0, lane);
        acc0 = wmma_bf16(a0, b, acc0);
        acc1 = wmma_bf16(a1, b, acc1);
    }
}

// Gate wave: accumulate i,f,g,o tiles (same 16 hidden units) for both 16-row batch halves.
// B-frag for gate g comes from weight rows g*Hd + utile*16 + (lane&15): 16 contiguous bf16.
DEVINL void gemm_acc8(const __bf16* A, int K, const __bf16* W, int utile, int lane,
                      v8f& i0, v8f& i1, v8f& f0, v8f& f1,
                      v8f& g0, v8f& g1, v8f& o0, v8f& o1) {
    const __bf16* wp = W + (size_t)(utile * 16 + (lane & 15)) * K + (lane >> 4) * 16;
    const size_t gs = (size_t)Hd * K;
    for (int k0 = 0; k0 < K; k0 += 32) {
        bf16x16 a0 = load_afrag(A, K, 0, k0, lane);
        bf16x16 a1 = load_afrag(A, K, 1, k0, lane);
        bf16x16 bi = *reinterpret_cast<const bf16x16*>(wp + k0);
        bf16x16 bf = *reinterpret_cast<const bf16x16*>(wp + gs + k0);
        bf16x16 bg = *reinterpret_cast<const bf16x16*>(wp + 2 * gs + k0);
        bf16x16 bo = *reinterpret_cast<const bf16x16*>(wp + 3 * gs + k0);
        i0 = wmma_bf16(a0, bi, i0);  i1 = wmma_bf16(a1, bi, i1);
        f0 = wmma_bf16(a0, bf, f0);  f1 = wmma_bf16(a1, bf, f1);
        g0 = wmma_bf16(a0, bg, g0);  g1 = wmma_bf16(a1, bg, g1);
        o0 = wmma_bf16(a0, bo, o0);  o1 = wmma_bf16(a1, bo, o1);
    }
}

// ---------------- device-wide sense-reversing barrier ----------------
DEVINL void grid_barrier(unsigned* bar) {
    __threadfence();
    __syncthreads();
    if (threadIdx.x == 0) {
        unsigned* cnt = bar;
        unsigned* gen = bar + 16;
        unsigned g = __hip_atomic_load(gen, __ATOMIC_RELAXED, __HIP_MEMORY_SCOPE_AGENT);
        unsigned a = __hip_atomic_fetch_add(cnt, 1u, __ATOMIC_ACQ_REL, __HIP_MEMORY_SCOPE_AGENT);
        if (a == (unsigned)NWG - 1u) {
            __hip_atomic_store(cnt, 0u, __ATOMIC_RELAXED, __HIP_MEMORY_SCOPE_AGENT);
            __hip_atomic_fetch_add(gen, 1u, __ATOMIC_RELEASE, __HIP_MEMORY_SCOPE_AGENT);
        } else {
            while (__hip_atomic_load(gen, __ATOMIC_ACQUIRE, __HIP_MEMORY_SCOPE_AGENT) == g) {
                __builtin_amdgcn_s_sleep(2);
            }
        }
    }
    __syncthreads();
}

// fast activations: v_exp_f32 + v_rcp_f32 (transcendental VALU co-executes with XDL)
DEVINL float sigf(float x)     { return __builtin_amdgcn_rcpf(1.0f + __expf(-x)); }
DEVINL float tanh_fast(float x){ return 2.0f * sigf(2.0f * x) - 1.0f; }

// ---------------- prep kernels ----------------
__global__ void k_zero(unsigned* p, size_t n) {
    for (size_t i = blockIdx.x * (size_t)blockDim.x + threadIdx.x; i < n;
         i += (size_t)gridDim.x * blockDim.x) p[i] = 0u;
}

__global__ void k_cvt_pad(__bf16* dst, const float* src, int dst_rows, int dst_cols,
                          int src_rows, int src_cols) {
    size_t n = (size_t)dst_rows * dst_cols;
    for (size_t i = blockIdx.x * (size_t)blockDim.x + threadIdx.x; i < n;
         i += (size_t)gridDim.x * blockDim.x) {
        int r = (int)(i / dst_cols), c = (int)(i % dst_cols);
        float v = (r < src_rows && c < src_cols) ? src[(size_t)r * src_cols + c] : 0.0f;
        dst[i] = (__bf16)v;
    }
}

__global__ void k_bsum(float* bsum,
                       const float* bi1, const float* bh1,
                       const float* bi2, const float* bh2,
                       const float* bi3, const float* bh3) {
    int i = blockIdx.x * blockDim.x + threadIdx.x;
    if (i >= 3 * G4H) return;
    int l = i / G4H, j = i % G4H;
    const float* bi = (l == 0) ? bi1 : (l == 1) ? bi2 : bi3;
    const float* bh = (l == 0) ? bh1 : (l == 1) ? bh2 : bh3;
    bsum[i] = bi[j] + bh[j];
}

__global__ void k_seed_x(__bf16* x0, const float* seq) {
    int i = blockIdx.x * blockDim.x + threadIdx.x;
    if (i >= B * F) return;
    int b = i / F, f = i % F;
    x0[b * KX + f] = (__bf16)seq[((size_t)b * SEED + 0) * F + f];
}

// ---------------- persistent fused LSTM rollout ----------------
// Iteration k (k = 0..T):
//   decoder waves : (k<SEED) seed-convert x[k] + flag FIRST, then decode t=k-1;
//                   (k>=SEED) decode t=k-1, write out + x[k], then flag.
//   gate waves    : gates+cell update for step t=k (k<T); layer0 spins on flag (>= 11*k)
//                   before its x-GEMM; c state lives in registers for the whole rollout.
//   grid barrier.
__global__ __launch_bounds__(BLOCK)
void k_lstm_persistent(char* ws, const float* seq, const int* ngen_p,
                       const float* bdec, float* out) {
    unsigned* bar  = (unsigned*)(ws + OFF_BAR);
    unsigned* flag = bar + 32;
    const __bf16* Wih1 = (const __bf16*)(ws + OFF_WIH1);
    const __bf16* Whh1 = (const __bf16*)(ws + OFF_WHH1);
    const __bf16* Wih2 = (const __bf16*)(ws + OFF_WIH2);
    const __bf16* Whh2 = (const __bf16*)(ws + OFF_WHH2);
    const __bf16* Wih3 = (const __bf16*)(ws + OFF_WIH3);
    const __bf16* Whh3 = (const __bf16*)(ws + OFF_WHH3);
    const __bf16* Wdec = (const __bf16*)(ws + OFF_WDEC);
    const float*  bsum = (const float*)(ws + OFF_BSUM);
    __bf16* hbuf = (__bf16*)(ws + OFF_H);
    __bf16* xbuf = (__bf16*)(ws + OFF_X);

    const int T = SEED + ngen_p[0];
    const size_t ostride = (size_t)T * F;

    const unsigned tid  = threadIdx.x;
    const unsigned gid  = blockIdx.x * BLOCK + tid;
    const unsigned wid  = gid >> 5;
    const int lane = (int)(tid & 31);
    const bool is_gate = wid < (unsigned)NGATEW;
    const bool is_dec  = wid >= (unsigned)DECW0 && wid < (unsigned)(DECW0 + NT_DEC);
    const int layer = is_gate ? (int)(wid >> 6) : 0;   // 0..2
    const int utile = is_gate ? (int)(wid & 63) : 0;   // 0..63
    const int dtile = (int)(wid - DECW0);              // decoder tile

    // per-lane constants
    const int colq = (lane & 15);
    const int rb   = (lane >> 4) * 8;
    float bI = 0.f, bFg = 0.f, bG = 0.f, bO = 0.f;
    if (is_gate) {
        const float* bs = bsum + (size_t)layer * G4H + utile * 16 + colq;
        bI = bs[0]; bFg = bs[Hd]; bG = bs[2 * Hd]; bO = bs[3 * Hd];
    }
    const int dcol = dtile * 16 + colq;
    const float bD = (is_dec && dcol < F) ? bdec[dcol] : 0.f;

    // persistent cell state in registers: rows rb+r (mtile0) and rb+r+16 (mtile1)
    float cst[16];
    #pragma unroll
    for (int r = 0; r < 16; ++r) cst[r] = 0.f;

    for (int k = 0; k <= T; ++k) {
        const int p  = k & 1;
        const int pn = p ^ 1;

        if (is_dec && k >= 1) {
            const int td = k - 1;
            const __bf16* h2n = hbuf + ((size_t)p * 3 + 2) * B * Hd;
            __bf16* xk = xbuf + (size_t)p * B * KX;   // input for step k

            if (k < SEED) {
                // teacher-forced x[k]: publish BEFORE the decoder GEMM so layer-0
                // waves are unblocked immediately during the whole seed phase.
                for (int i = dtile * 32 + lane; i < B * F; i += NT_DEC * 32) {
                    int b = i / F, f = i % F;
                    xk[(size_t)b * KX + f] = (__bf16)seq[((size_t)b * SEED + k) * F + f];
                }
                __threadfence();
                if (lane == 0)
                    __hip_atomic_fetch_add(flag, 1u, __ATOMIC_RELEASE, __HIP_MEMORY_SCOPE_AGENT);
            }

            // ---- decode step t=k-1; h2_new of that step sits at parity k&1 ----
            v8f d0 = {}; v8f d1 = {};
            gemm_acc2(h2n, Hd, Wdec, dtile, lane, d0, d1);
            if (dcol < F) {
                #pragma unroll
                for (int r = 0; r < 8; ++r) {
                    float v0 = d0[r] + bD, v1 = d1[r] + bD;
                    out[(size_t)(rb + r) * ostride + (size_t)td * F + dcol]      = v0;
                    out[(size_t)(rb + r + 16) * ostride + (size_t)td * F + dcol] = v1;
                    if (k >= SEED) {   // autoregressive feedback becomes x[k]
                        xk[(size_t)(rb + r) * KX + dcol]      = (__bf16)v0;
                        xk[(size_t)(rb + r + 16) * KX + dcol] = (__bf16)v1;
                    }
                }
            }
            if (k >= SEED) {
                __threadfence();
                if (lane == 0)
                    __hip_atomic_fetch_add(flag, 1u, __ATOMIC_RELEASE, __HIP_MEMORY_SCOPE_AGENT);
            }
        }

        if (is_gate && k < T) {
            // ---- gates for step t=k: i,f,g,o tiles of this wave's 16 units ----
            const __bf16* xc = xbuf + (size_t)p * B * KX;
            const __bf16* h0 = hbuf + ((size_t)p * 3 + 0) * B * Hd;
            const __bf16* h1 = hbuf + ((size_t)p * 3 + 1) * B * Hd;
            const __bf16* h2 = hbuf + ((size_t)p * 3 + 2) * B * Hd;
            v8f i0 = {}; v8f i1 = {}; v8f f0 = {}; v8f f1 = {};
            v8f g0 = {}; v8f g1 = {}; v8f o0 = {}; v8f o1 = {};
            if (layer == 0) {
                gemm_acc8(h0, Hd, Whh1, utile, lane, i0, i1, f0, f1, g0, g1, o0, o1);
                // wait for this step's input frame (decoder of step k-1 / seed writer)
                const unsigned target = 11u * (unsigned)k;
                while (__hip_atomic_load(flag, __ATOMIC_ACQUIRE, __HIP_MEMORY_SCOPE_AGENT) < target)
                    __builtin_amdgcn_s_sleep(1);
                gemm_acc8(xc, KX, Wih1, utile, lane, i0, i1, f0, f1, g0, g1, o0, o1);
            } else if (layer == 1) {
                gemm_acc8(h0, Hd, Wih2, utile, lane, i0, i1, f0, f1, g0, g1, o0, o1); // stale h0
                gemm_acc8(h1, Hd, Whh2, utile, lane, i0, i1, f0, f1, g0, g1, o0, o1);
            } else {
                gemm_acc8(h1, Hd, Wih3, utile, lane, i0, i1, f0, f1, g0, g1, o0, o1); // stale h1
                gemm_acc8(h2, Hd, Whh3, utile, lane, i0, i1, f0, f1, g0, g1, o0, o1);
            }
            // ---- fused cell update in registers; write h_new[pn] only ----
            __bf16* hdst = hbuf + ((size_t)pn * 3 + layer) * B * Hd;
            const int col = utile * 16 + colq;
            #pragma unroll
            for (int r = 0; r < 8; ++r) {
                float cn0 = sigf(f0[r] + bFg) * cst[r] + sigf(i0[r] + bI) * tanh_fast(g0[r] + bG);
                cst[r] = cn0;
                hdst[(size_t)(rb + r) * Hd + col] = (__bf16)(sigf(o0[r] + bO) * tanh_fast(cn0));
                float cn1 = sigf(f1[r] + bFg) * cst[8 + r] + sigf(i1[r] + bI) * tanh_fast(g1[r] + bG);
                cst[8 + r] = cn1;
                hdst[(size_t)(rb + r + 16) * Hd + col] = (__bf16)(sigf(o1[r] + bO) * tanh_fast(cn1));
            }
        }

        grid_barrier(bar);
    }
}

// ---------------- launch ----------------
extern "C" void kernel_launch(void* const* d_in, const int* in_sizes, int n_in,
                              void* d_out, int out_size, void* d_ws, size_t ws_size,
                              hipStream_t stream) {
    (void)in_sizes; (void)n_in; (void)out_size; (void)ws_size;
    const float* seq  = (const float*)d_in[0];
    const int*   ngen = (const int*)d_in[1];
    const float* Wih1 = (const float*)d_in[2];
    const float* Whh1 = (const float*)d_in[3];
    const float* bih1 = (const float*)d_in[4];
    const float* bhh1 = (const float*)d_in[5];
    const float* Wih2 = (const float*)d_in[6];
    const float* Whh2 = (const float*)d_in[7];
    const float* bih2 = (const float*)d_in[8];
    const float* bhh2 = (const float*)d_in[9];
    const float* Wih3 = (const float*)d_in[10];
    const float* Whh3 = (const float*)d_in[11];
    const float* bih3 = (const float*)d_in[12];
    const float* bhh3 = (const float*)d_in[13];
    const float* Wdec = (const float*)d_in[14];
    const float* bdec = (const float*)d_in[15];
    char* ws = (char*)d_ws;

    // zero barrier/flag + h/x state (cell state lives in registers now)
    k_zero<<<1, 64, 0, stream>>>((unsigned*)(ws + OFF_BAR), 64);
    k_zero<<<256, 256, 0, stream>>>((unsigned*)(ws + OFF_H), (OFF_X + SZ_X - OFF_H) / 4);

    // weights -> bf16 once per launch (~131 MB of traffic, ~6 us at HBM rate)
    k_cvt_pad<<<1024, 256, 0, stream>>>((__bf16*)(ws + OFF_WIH1), Wih1, G4H, KX, G4H, F);
    k_cvt_pad<<<2048, 256, 0, stream>>>((__bf16*)(ws + OFF_WHH1), Whh1, G4H, Hd, G4H, Hd);
    k_cvt_pad<<<2048, 256, 0, stream>>>((__bf16*)(ws + OFF_WIH2), Wih2, G4H, Hd, G4H, Hd);
    k_cvt_pad<<<2048, 256, 0, stream>>>((__bf16*)(ws + OFF_WHH2), Whh2, G4H, Hd, G4H, Hd);
    k_cvt_pad<<<2048, 256, 0, stream>>>((__bf16*)(ws + OFF_WIH3), Wih3, G4H, Hd, G4H, Hd);
    k_cvt_pad<<<2048, 256, 0, stream>>>((__bf16*)(ws + OFF_WHH3), Whh3, G4H, Hd, G4H, Hd);
    k_cvt_pad<<<256, 256, 0, stream>>>((__bf16*)(ws + OFF_WDEC), Wdec, 176, Hd, F, Hd);

    k_bsum<<<(3 * G4H + 255) / 256, 256, 0, stream>>>((float*)(ws + OFF_BSUM),
                                                      bih1, bhh1, bih2, bhh2, bih3, bhh3);
    k_seed_x<<<(B * F + 255) / 256, 256, 0, stream>>>((__bf16*)(ws + OFF_X), seq);

    // single persistent kernel: 500 timesteps, 1 grid barrier + 1 flag handshake per step
    k_lstm_persistent<<<NWG, BLOCK, 0, stream>>>(ws, seq, ngen, bdec, (float*)d_out);
}
